// OnlineCNNExtractor_1786706395845
// MI455X (gfx1250) — matrix operand order repaired and checked by reading, not compiled, not run on hardware
//
#include <hip/hip_runtime.h>
#include <stddef.h>

// ---------------------------------------------------------------------------
// Types / helpers
// ---------------------------------------------------------------------------
typedef __attribute__((ext_vector_type(16))) __bf16 v16bf;
typedef __attribute__((ext_vector_type(8)))  float  v8f;
typedef __attribute__((ext_vector_type(4)))  float  v4f;
typedef __attribute__((ext_vector_type(4)))  unsigned int u32x4;
typedef __attribute__((ext_vector_type(8)))  int    i32x8;
typedef __attribute__((ext_vector_type(4)))  int    i32x4;

__device__ __forceinline__ unsigned short f2bf(float f) {
  unsigned int x = __builtin_bit_cast(unsigned int, f);
  x += 0x7fffu + ((x >> 16) & 1u);          // round-to-nearest-even
  return (unsigned short)(x >> 16);
}
__device__ __forceinline__ float bf2f(unsigned short u) {
  unsigned int x = ((unsigned int)u) << 16;
  return __builtin_bit_cast(float, x);
}
__device__ __forceinline__ v16bf pack_bf(uint4 q0, uint4 q1) {
  union { uint4 q[2]; v16bf v; } u;
  u.q[0] = q0; u.q[1] = q1;
  return u.v;
}
__device__ __forceinline__ uint4 zsel(bool c, uint4 v) {
  uint4 r;
  r.x = c ? v.x : 0u; r.y = c ? v.y : 0u;
  r.z = c ? v.z : 0u; r.w = c ? v.w : 0u;
  return r;
}

// ---------------------------------------------------------------------------
// Elementwise f32 -> bf16 (proj_w)
// ---------------------------------------------------------------------------
__global__ void f32_to_bf16_kernel(const float* __restrict__ in,
                                   unsigned short* __restrict__ out, long n) {
  long i = (long)blockIdx.x * blockDim.x + threadIdx.x;
  long st = (long)gridDim.x * blockDim.x;
  for (; i < n; i += st) out[i] = f2bf(in[i]);
}

// ---------------------------------------------------------------------------
// Repack OIHW f32 conv weights -> [O][kh*kw][I] bf16 (I % 32 == 0)
// ---------------------------------------------------------------------------
__global__ void repack_w_kernel(const float* __restrict__ src,
                                unsigned short* __restrict__ dst,
                                int O, int I, int khw) {
  long total = (long)O * I * khw;
  long i = (long)blockIdx.x * blockDim.x + threadIdx.x;
  long st = (long)gridDim.x * blockDim.x;
  for (; i < total; i += st) {
    int o = (int)(i / ((long)khw * I));
    int rr = (int)(i % ((long)khw * I));
    int cell = rr / I;
    int ci = rr - cell * I;
    dst[i] = f2bf(src[((size_t)o * I + ci) * khw + cell]);
  }
}

// ---------------------------------------------------------------------------
// conv1 weight pack: [64,3,7,7] -> [64,160] (k = cell*3+ci, zero padded)
// ---------------------------------------------------------------------------
__global__ void pack_conv1_w_kernel(const float* __restrict__ src,
                                    unsigned short* __restrict__ dst) {
  long i = (long)blockIdx.x * blockDim.x + threadIdx.x;
  long st = (long)gridDim.x * blockDim.x;
  for (; i < 64L * 160; i += st) {
    int o = (int)(i / 160), k = (int)(i % 160);
    unsigned short v = 0;
    if (k < 147) {
      int cell = k / 3, ci = k - cell * 3;
      v = f2bf(src[((size_t)o * 3 + ci) * 49 + cell]);
    }
    dst[i] = v;
  }
}

// ---------------------------------------------------------------------------
// im2col for conv1: raw [16,3,224,224] f32 -> col [200704,160] bf16
// ---------------------------------------------------------------------------
__global__ void im2col_conv1_kernel(const float* __restrict__ raw,
                                    unsigned short* __restrict__ col) {
  const long total = 200704L * 160;
  long i = (long)blockIdx.x * blockDim.x + threadIdx.x;
  long st = (long)gridDim.x * blockDim.x;
  for (; i < total; i += st) {
    int k = (int)(i % 160);
    long n = i / 160;
    int img = (int)(n / 12544);
    int r = (int)(n % 12544);
    int oy = r / 112, ox = r % 112;
    unsigned short v = 0;
    if (k < 147) {
      int cell = k / 3, ci = k - cell * 3;
      int ky = cell / 7, kx = cell % 7;
      int iy = oy * 2 - 3 + ky, ix = ox * 2 - 3 + kx;
      if (iy >= 0 && iy < 224 && ix >= 0 && ix < 224)
        v = f2bf(raw[(((size_t)img * 3 + ci) * 224 + iy) * 224 + ix]);
    }
    col[i] = v;
  }
}

// ---------------------------------------------------------------------------
// Generic WMMA GEMM:  out[n][m] = sum_k A[m][k] * B[n][k]  (+ bias[m])
// A [M,K] bf16, B [Nrows,K] bf16, both row-major.
// The 32xK A panel is DMA'd into LDS by the Tensor Data Mover (one
// tensor_load_to_lds per block, issued by wave 0, tracked by TENSORcnt);
// each wave computes 32 M x 16 N (2 accumulators) so one B fragment feeds
// 2 wmma.  K % 32 == 0, K <= 512, M % 32 == 0.
// ---------------------------------------------------------------------------
__global__ void __launch_bounds__(128)
gemm_wmma(const unsigned short* __restrict__ A,
          const unsigned short* __restrict__ B,
          const float* __restrict__ bias,
          float* __restrict__ out,
          int M, int Nrows, int K) {
  __shared__ unsigned short sA[32 * 512];       // 32 KB panel
  const int lane = threadIdx.x & 31;
  const int wave = threadIdx.x >> 5;
  const int tileN = blockIdx.x * 4 + wave;
  const int tileM = blockIdx.y;                 // 32-wide panel, shared by block
  const int half = lane >> 4, l16 = lane & 15;

  // --- TDM: DMA the [32][K] bf16 panel global -> LDS (wave 0 only) ---
  if (wave == 0) {
    unsigned long long ga =
        (unsigned long long)(const void*)(A + (size_t)tileM * 32 * K);
    unsigned ldsa = (unsigned)(unsigned long long)(void*)sA;  // LDS offset
    // D# group 0: count=1 | lds_addr | global_addr[56:0] | type=2
    u32x4 g0 = {1u, ldsa, (unsigned)ga,
                (unsigned)((ga >> 32) & 0x1FFFFFFull) | (2u << 30)};
    // D# group 1: data_size=2B; tensor_dim0=K, tensor_dim1=32,
    //             tile_dim0=K, tile_dim1=32, dim0_stride=K
    i32x8 g1 = {(int)(1u << 16),
                (int)(((unsigned)K & 0xFFFFu) << 16),
                (int)((((unsigned)K >> 16) & 0xFFFFu) | (32u << 16)),
                (int)(((unsigned)K & 0xFFFFu) << 16),
                32, (int)K, 0, 0};
    i32x4 g2 = {0, 0, 0, 0};
    i32x4 g3 = {0, 0, 0, 0};
    i32x8 g4 = {0, 0, 0, 0, 0, 0, 0, 0};
    __builtin_amdgcn_tensor_load_to_lds(g0, g1, g2, g3, g4, 0);
    __builtin_amdgcn_s_wait_tensorcnt(0);
  }
  __syncthreads();

  if (tileN * 16 >= Nrows) return;              // wave-uniform

  int nIdx = tileN * 16 + l16;
  const bool nValid = nIdx < Nrows;
  if (!nValid) nIdx = Nrows - 1;
  const unsigned short* brow = B + (size_t)nIdx * K + half * 8;
  const unsigned short* arow0 = sA + (size_t)l16 * K + half * 8;        // rows 0..15
  const unsigned short* arow1 = sA + (size_t)(16 + l16) * K + half * 8; // rows 16..31

  v8f acc0 = {0.f, 0.f, 0.f, 0.f, 0.f, 0.f, 0.f, 0.f};
  v8f acc1 = acc0;
  for (int kb = 0; kb < K; kb += 32) {
    uint4 b0 = *(const uint4*)(brow + kb);       // global_load_b128
    uint4 b1 = *(const uint4*)(brow + kb + 16);
    v16bf bf = pack_bf(b0, b1);
    uint4 a00 = *(const uint4*)(arow0 + kb);     // ds_load_b128
    uint4 a01 = *(const uint4*)(arow0 + kb + 16);
    uint4 a10 = *(const uint4*)(arow1 + kb);
    uint4 a11 = *(const uint4*)(arow1 + kb + 16);
    acc0 = __builtin_amdgcn_wmma_f32_16x16x32_bf16(
        false, pack_bf(a00, a01), false, bf, (short)0, acc0, false, false);
    acc1 = __builtin_amdgcn_wmma_f32_16x16x32_bf16(
        false, pack_bf(a10, a11), false, bf, (short)0, acc1, false, false);
  }

  if (nValid) {
    float* orow = out + (size_t)nIdx * M + tileM * 32;
#pragma unroll
    for (int s = 0; s < 2; ++s) {
      v8f acc = s ? acc1 : acc0;
      int m0 = s * 16 + half * 8;
      v4f lo = __builtin_shufflevector(acc, acc, 0, 1, 2, 3);
      v4f hi = __builtin_shufflevector(acc, acc, 4, 5, 6, 7);
      if (bias) {
        lo += *(const v4f*)(bias + tileM * 32 + m0);
        hi += *(const v4f*)(bias + tileM * 32 + m0 + 4);
      }
      *(v4f*)(orow + m0) = lo;
      *(v4f*)(orow + m0 + 4) = hi;
    }
  }
}

// ---------------------------------------------------------------------------
// NHWC implicit-GEMM conv via v_wmma_f32_16x16x32_bf16.
// in [N,Hin,Win,C] bf16 (C % 32 == 0), w [Cout][kh*kw][C] bf16,
// out [N,Hout,Wout,Cout] f32 (Cout % 64 == 0).
// Each wave computes 64 Cout x 16 pixels (4 accumulators): one B (pixel)
// fragment feeds 4 wmma. All fragment loads are b128; padding handled with
// clamped addresses + cndmask zeroing (no divergent branches).
// ---------------------------------------------------------------------------
__global__ void __launch_bounds__(128)
conv_nhwc_wmma(const unsigned short* __restrict__ in,
               const unsigned short* __restrict__ w,
               float* __restrict__ out,
               int N, int C, int Hin, int Win,
               int Cout, int Hout, int Wout,
               int kh, int kw, int stride, int pad) {
  const int lane = threadIdx.x & 31;
  const int wave = threadIdx.x >> 5;
  const int tileN = blockIdx.x * 4 + wave;
  const int tileM = blockIdx.y;                 // 64 output channels
  const int ntot = N * Hout * Wout;
  if (tileN * 16 >= ntot) return;               // wave-uniform

  const int half = lane >> 4, l16 = lane & 15;
  const int khw = kh * kw;
  const int K = khw * C;

  int nIdx = tileN * 16 + l16;
  const bool nValid = nIdx < ntot;
  if (!nValid) nIdx = ntot - 1;
  const int img = nIdx / (Hout * Wout);
  int r = nIdx % (Hout * Wout);
  const int oy = r / Wout, ox = r % Wout;
  const int iy0 = oy * stride - pad;
  const int ix0 = ox * stride - pad;

  const unsigned short* wrow[4];
#pragma unroll
  for (int s = 0; s < 4; ++s)
    wrow[s] = w + (size_t)(tileM * 64 + s * 16 + l16) * K + half * 8;

  v8f acc[4];
#pragma unroll
  for (int s = 0; s < 4; ++s)
    acc[s] = (v8f){0.f, 0.f, 0.f, 0.f, 0.f, 0.f, 0.f, 0.f};

  for (int cell = 0; cell < khw; ++cell) {
    const int ky = cell / kw, kx = cell - (cell / kw) * kw;
    const int iy = iy0 + ky, ix = ix0 + kx;
    const bool inb = (iy >= 0) & (iy < Hin) & (ix >= 0) & (ix < Win);
    int iyc = iy < 0 ? 0 : (iy >= Hin ? Hin - 1 : iy);
    int ixc = ix < 0 ? 0 : (ix >= Win ? Win - 1 : ix);
    const unsigned short* brow =
        in + (((size_t)img * Hin + iyc) * Win + ixc) * C + half * 8;
    const size_t cbase = (size_t)cell * C;
    for (int cb = 0; cb < C; cb += 32) {
      uint4 b0 = zsel(inb, *(const uint4*)(brow + cb));
      uint4 b1 = zsel(inb, *(const uint4*)(brow + cb + 16));
      v16bf bf = pack_bf(b0, b1);
#pragma unroll
      for (int s = 0; s < 4; ++s) {
        uint4 a0 = *(const uint4*)(wrow[s] + cbase + cb);
        uint4 a1 = *(const uint4*)(wrow[s] + cbase + cb + 16);
        acc[s] = __builtin_amdgcn_wmma_f32_16x16x32_bf16(
            false, pack_bf(a0, a1), false, bf, (short)0, acc[s], false, false);
      }
    }
  }

  if (nValid) {
    float* orow = out + (size_t)nIdx * Cout + tileM * 64;
#pragma unroll
    for (int s = 0; s < 4; ++s) {
      int m0 = s * 16 + half * 8;
      *(v4f*)(orow + m0) =
          __builtin_shufflevector(acc[s], acc[s], 0, 1, 2, 3);
      *(v4f*)(orow + m0 + 4) =
          __builtin_shufflevector(acc[s], acc[s], 4, 5, 6, 7);
    }
  }
}

// ---------------------------------------------------------------------------
// Fused BN (+skip) (+ReLU), NHWC: channel = i & (C-1)  (C is a power of two).
// ---------------------------------------------------------------------------
__global__ void bn_act_kernel(const float* __restrict__ x,
                              const float* __restrict__ scale,
                              const float* __restrict__ bias,
                              const unsigned short* __restrict__ skip_bf,
                              const float* __restrict__ skip_f,
                              unsigned short* __restrict__ out_bf,
                              float* __restrict__ out_f,
                              int Cmask, long total, int relu) {
  long i = (long)blockIdx.x * blockDim.x + threadIdx.x;
  long st = (long)gridDim.x * blockDim.x;
  for (; i < total; i += st) {
    int c = (int)i & Cmask;
    float v = x[i] * scale[c] + bias[c];
    if (skip_bf) v += bf2f(skip_bf[i]);
    if (skip_f)  v += skip_f[i];
    if (relu)    v = fmaxf(v, 0.f);
    if (out_bf)  out_bf[i] = f2bf(v);
    if (out_f)   out_f[i]  = v;
  }
}

// ---------------------------------------------------------------------------
// 3x3 / stride-2 / pad-1 max pool, NHWC bf16 -> bf16 (C = 64)
// ---------------------------------------------------------------------------
__global__ void maxpool_nhwc_kernel(const unsigned short* __restrict__ in,
                                    unsigned short* __restrict__ out,
                                    int N, int C, int Hin, int Win,
                                    int Hout, int Wout) {
  long total = (long)N * Hout * Wout * C;
  long i = (long)blockIdx.x * blockDim.x + threadIdx.x;
  long st = (long)gridDim.x * blockDim.x;
  for (; i < total; i += st) {
    int c  = (int)(i % C);
    int ox = (int)((i / C) % Wout);
    int oy = (int)((i / ((long)C * Wout)) % Hout);
    int n  = (int)(i / ((long)C * Wout * Hout));
    float m = -3.402823466e38f;
    for (int ky = 0; ky < 3; ++ky) {
      int iy = oy * 2 - 1 + ky;
      if (iy < 0 || iy >= Hin) continue;
      for (int kx = 0; kx < 3; ++kx) {
        int ix = ox * 2 - 1 + kx;
        if (ix < 0 || ix >= Win) continue;
        m = fmaxf(m, bf2f(in[(((size_t)n * Hin + iy) * Win + ix) * C + c]));
      }
    }
    out[i] = f2bf(m);
  }
}

// ---------------------------------------------------------------------------
// Bilinear grid-sample fallback, feat NHWC [16,7,7,512] -> fb [16,256,512]
// ---------------------------------------------------------------------------
__global__ void grid_sample_kernel(const float* __restrict__ feat,
                                   const float* __restrict__ pos,
                                   float* __restrict__ fb) {
  int bn = blockIdx.x;                 // b*256 + node
  int b  = bn >> 8;
  float gx = pos[(size_t)bn * 2 + 0];
  float gy = pos[(size_t)bn * 2 + 1];
  float x = fminf(fmaxf(gx * 6.f, 0.f), 6.f);
  float y = fminf(fmaxf(gy * 6.f, 0.f), 6.f);
  float x0f = floorf(x), y0f = floorf(y);
  int x0 = (int)x0f; x0 = x0 < 0 ? 0 : (x0 > 6 ? 6 : x0);
  int y0 = (int)y0f; y0 = y0 < 0 ? 0 : (y0 > 6 ? 6 : y0);
  int x1 = x0 + 1 > 6 ? 6 : x0 + 1;
  int y1 = y0 + 1 > 6 ? 6 : y0 + 1;
  float wx = x - x0f, wy = y - y0f;
  const float* fbase = feat + (size_t)b * 49 * 512;
  for (int c = threadIdx.x; c < 512; c += blockDim.x) {
    float v00 = fbase[(y0 * 7 + x0) * 512 + c];
    float v01 = fbase[(y0 * 7 + x1) * 512 + c];
    float v10 = fbase[(y1 * 7 + x0) * 512 + c];
    float v11 = fbase[(y1 * 7 + x1) * 512 + c];
    fb[(size_t)bn * 512 + c] =
        v00 * (1 - wx) * (1 - wy) + v01 * wx * (1 - wy) +
        v10 * (1 - wx) * wy + v11 * wx * wy;
  }
}

// ---------------------------------------------------------------------------
// Segment pooling over 7x7 subsampled mask, feat NHWC; writes pooled bf16.
// ---------------------------------------------------------------------------
__global__ void segpool_kernel(const float* __restrict__ feat,   // [16,49,512]
                               const int* __restrict__ mask,     // [16,224,224]
                               const float* __restrict__ fb,     // [16,256,512]
                               unsigned short* __restrict__ pooled) {
  __shared__ int   sms[49];
  __shared__ float scnt;
  int bn = blockIdx.x;                 // b*256 + node
  int b = bn >> 8, node = bn & 255;
  int tid = threadIdx.x;
  if (tid < 49) {
    int py = tid / 7, px = tid % 7;    // mask[:, ::32, ::32]
    sms[tid] = mask[(size_t)b * 224 * 224 + (size_t)py * 32 * 224 + px * 32];
  }
  __syncthreads();
  if (tid == 0) {
    int c = 0;
    for (int p = 0; p < 49; ++p) c += (sms[p] == node);
    scnt = (float)c;
  }
  __syncthreads();
  float cnt = scnt;
  const float* fbase = feat + (size_t)b * 49 * 512;
  for (int c = tid; c < 512; c += blockDim.x) {
    float sum = 0.f;
    for (int p = 0; p < 49; ++p)
      if (sms[p] == node) sum += fbase[p * 512 + c];
    float v = (cnt > 0.f) ? (sum / cnt) : fb[(size_t)bn * 512 + c];
    pooled[(size_t)bn * 512 + c] = f2bf(v);
  }
}

// ---------------------------------------------------------------------------
// Host orchestration
// ---------------------------------------------------------------------------
extern "C" void kernel_launch(void* const* d_in, const int* in_sizes, int n_in,
                              void* d_out, int out_size, void* d_ws, size_t ws_size,
                              hipStream_t stream) {
  (void)in_sizes; (void)n_in; (void)out_size; (void)ws_size;
  auto F = [&](int i) { return (const float*)d_in[i]; };
  const float* raw  = F(0);                  // [16,3,224,224]
  const float* pos  = F(1);                  // [16,256,2]
  const int*   mask = (const int*)d_in[2];   // [16,224,224]

  // Param leaves (JAX pytree sorted-key flattening of the params dict):
  // 3:bn1_b 4:bn1_s 5:conv1_w, blocks: bn1b,bn1s,bn2b,bn2s,c1w,c2w[,dnb,dns,dww]
  struct BlkDesc { int idx, cin, cout, hin, stride; bool dw; };
  const BlkDesc blks[8] = {
      { 6,  64,  64, 56, 1, false},   // l1b0
      {12,  64,  64, 56, 1, false},   // l1b1
      {18,  64, 128, 56, 2, true },   // l2b0
      {27, 128, 128, 28, 1, false},   // l2b1
      {33, 128, 256, 28, 2, true },   // l3b0
      {42, 256, 256, 14, 1, false},   // l3b1
      {48, 256, 512, 14, 2, true },   // l4b0
      {57, 512, 512,  7, 1, false},   // l4b1
  };
  const int I_PROJ_B = 63, I_PROJ_W = 64;

  // Workspace bump allocator
  char* ws = (char*)d_ws;
  size_t off = 0;
  auto alloc = [&](size_t bytes) -> void* {
    void* p = ws + off;
    off = (off + bytes + 255) & ~(size_t)255;
    return p;
  };

  unsigned short* w_conv1 = (unsigned short*)alloc((size_t)64 * 160 * 2);
  unsigned short *wc1p[8], *wc2p[8], *wdwp[8];
  for (int i = 0; i < 8; ++i) {
    wc1p[i] = (unsigned short*)alloc((size_t)blks[i].cout * blks[i].cin * 9 * 2);
    wc2p[i] = (unsigned short*)alloc((size_t)blks[i].cout * blks[i].cout * 9 * 2);
    wdwp[i] = blks[i].dw
        ? (unsigned short*)alloc((size_t)blks[i].cout * blks[i].cin * 2)
        : nullptr;
  }
  unsigned short* w_proj = (unsigned short*)alloc((size_t)128 * 512 * 2);
  unsigned short* col1   = (unsigned short*)alloc((size_t)200704 * 160 * 2);
  unsigned short* bufA   = (unsigned short*)alloc((size_t)16 * 112 * 112 * 64 * 2);
  unsigned short* bufB   = (unsigned short*)alloc((size_t)16 * 112 * 112 * 64 * 2);
  float* t_f32  = (float*)alloc((size_t)16 * 112 * 112 * 64 * 4);
  float* sc_f32 = (float*)alloc((size_t)16 * 28 * 28 * 128 * 4);
  float* feat   = (float*)alloc((size_t)16 * 49 * 512 * 4);
  float* fb     = (float*)alloc((size_t)16 * 256 * 512 * 4);
  unsigned short* pooled = (unsigned short*)alloc((size_t)4096 * 512 * 2);

  auto bnact = [&](const float* x, const float* s, const float* b,
                   const unsigned short* skbf, const float* skf,
                   unsigned short* obf, float* of, int C, long total, int relu) {
    bn_act_kernel<<<4096, 256, 0, stream>>>(x, s, b, skbf, skf, obf, of,
                                            C - 1, total, relu);
  };
  auto conv = [&](const unsigned short* x, const unsigned short* wt, float* o,
                  int Cin, int Hin, int Win, int Cout, int Hout, int Wout,
                  int kh, int kw, int st, int pad) {
    int ntot = 16 * Hout * Wout;
    int tilesN = (ntot + 15) / 16;
    dim3 grid((tilesN + 3) / 4, Cout / 64);
    conv_nhwc_wmma<<<grid, 128, 0, stream>>>(x, wt, o, 16, Cin, Hin, Win,
                                             Cout, Hout, Wout, kh, kw, st, pad);
  };
  auto gemm = [&](const unsigned short* A, const unsigned short* B,
                  const float* bias, float* o, int M, int Nrows, int K) {
    int tilesN = (Nrows + 15) / 16;
    dim3 grid((tilesN + 3) / 4, M / 32);
    gemm_wmma<<<grid, 128, 0, stream>>>(A, B, bias, o, M, Nrows, K);
  };

  // --- weight packing ---
  pack_conv1_w_kernel<<<40, 256, 0, stream>>>(F(5), w_conv1);
  for (int i = 0; i < 8; ++i) {
    const BlkDesc& bd = blks[i];
    repack_w_kernel<<<2048, 256, 0, stream>>>(F(bd.idx + 4), wc1p[i],
                                              bd.cout, bd.cin, 9);
    repack_w_kernel<<<2048, 256, 0, stream>>>(F(bd.idx + 5), wc2p[i],
                                              bd.cout, bd.cout, 9);
    if (bd.dw)
      repack_w_kernel<<<1024, 256, 0, stream>>>(F(bd.idx + 8), wdwp[i],
                                                bd.cout, bd.cin, 1);
  }
  f32_to_bf16_kernel<<<256, 256, 0, stream>>>(F(I_PROJ_W), w_proj, 128L * 512);

  // --- stem: conv7x7 s2 (im2col + WMMA GEMM) + BN + ReLU + maxpool ---
  im2col_conv1_kernel<<<4096, 256, 0, stream>>>(raw, col1);
  gemm(w_conv1, col1, nullptr, t_f32, 64, 200704, 160);
  bnact(t_f32, F(4), F(3), nullptr, nullptr, bufA, nullptr,
        64, 200704L * 64, 1);
  maxpool_nhwc_kernel<<<4096, 256, 0, stream>>>(bufA, bufB,
                                                16, 64, 112, 112, 56, 56);
  unsigned short* cur = bufB;
  unsigned short* oth = bufA;

  // --- 8 basic blocks (NHWC) ---
  for (int i = 0; i < 8; ++i) {
    const BlkDesc& bd = blks[i];
    int hout = (bd.stride == 1) ? bd.hin : bd.hin / 2;
    long hw = (long)hout * hout;
    conv(cur, wc1p[i], t_f32, bd.cin, bd.hin, bd.hin,
         bd.cout, hout, hout, 3, 3, bd.stride, 1);
    bnact(t_f32, F(bd.idx + 1), F(bd.idx + 0), nullptr, nullptr,
          oth, nullptr, bd.cout, 16 * hw * bd.cout, 1);
    conv(oth, wc2p[i], t_f32, bd.cout, hout, hout,
         bd.cout, hout, hout, 3, 3, 1, 1);
    const unsigned short* skbf = nullptr;
    const float* skf = nullptr;
    if (bd.dw) {
      conv(cur, wdwp[i], sc_f32, bd.cin, bd.hin, bd.hin,
           bd.cout, hout, hout, 1, 1, bd.stride, 0);
      bnact(sc_f32, F(bd.idx + 7), F(bd.idx + 6), nullptr, nullptr,
            nullptr, sc_f32, bd.cout, 16 * hw * bd.cout, 0);
      skf = sc_f32;
    } else {
      skbf = cur;
    }
    float* of = (i == 7) ? feat : nullptr;
    bnact(t_f32, F(bd.idx + 3), F(bd.idx + 2), skbf, skf,
          oth, of, bd.cout, 16 * hw * bd.cout, 1);
    unsigned short* tmp = cur; cur = oth; oth = tmp;
  }

  // --- pooling + projection ---
  grid_sample_kernel<<<4096, 128, 0, stream>>>(feat, pos, fb);
  segpool_kernel<<<4096, 128, 0, stream>>>(feat, mask, fb, pooled);
  gemm(w_proj, pooled, F(I_PROJ_B), (float*)d_out, 128, 4096, 512);
}
